// MeshGraphNetContact_87608742903953
// MI455X (gfx1250) — compile-verified, matrix operand order checked by reading, not run on hardware
//
#include <hip/hip_runtime.h>
#include <hip/hip_bf16.h>
#include <stdint.h>

// ---------------------------------------------------------------------------
// MeshGraphNet-with-contact, fused bf16-WMMA implementation for gfx1250.
//   - fused gather + (384->128 GEMM, ReLU) + (128->128 GEMM) + LayerNorm +
//     residual in one kernel (modes: edge / node / encoder)
//   - scatter-mean implemented as deterministic CSR gather fused into the
//     node-MLP input stage (no float atomics)
//   - weights pre-packed into WMMA B-fragment-major bf16 layout so every
//     B-fragment load is a fully coalesced contiguous 1KB wave read (L2-hot)
// ---------------------------------------------------------------------------

typedef __attribute__((ext_vector_type(16))) __bf16 v16bf;
typedef __attribute__((ext_vector_type(8)))  float  v8f;

#define WG 256

union BFrag { uint4 u[2]; v16bf v; };

// A-fragment from LDS: 16B at +0 (K k0..k0+7) and 16B at +32B (K k0+16..k0+23)
__device__ __forceinline__ v16bf ld_frag_a(const __bf16* p) {
  BFrag f;
  f.u[0] = *(const uint4*)(p);
  f.u[1] = *(const uint4*)(p + 16);   // +16 elements = +32 bytes
  return f.v;
}
// B-fragment from packed global weights: 32 contiguous bytes per lane
__device__ __forceinline__ v16bf ld_frag_b(const __bf16* p) {
  BFrag f;
  f.u[0] = ((const uint4*)p)[0];
  f.u[1] = ((const uint4*)p)[1];
  return f.v;
}

// ---------------------------------------------------------------------------
// CSR construction kernels (deterministic aggregation order)
// ---------------------------------------------------------------------------
__global__ __launch_bounds__(256) void count_kernel(const int* __restrict__ r,
                                                    int* __restrict__ cnt, int ne) {
  int e = blockIdx.x * 256 + threadIdx.x;
  if (e < ne) atomicAdd(&cnt[r[e]], 1);
}

__global__ __launch_bounds__(1024) void scan_kernel(const int* __restrict__ cnt,
                                                    int* __restrict__ off,
                                                    int* __restrict__ cur,
                                                    int n, int chunk) {
  __shared__ int s[1024];
  int t = threadIdx.x;
  int lo = t * chunk;
  int sum = 0;
  for (int i = 0; i < chunk; ++i) { int idx = lo + i; if (idx < n) sum += cnt[idx]; }
  s[t] = sum;
  __syncthreads();
  for (int d = 1; d < 1024; d <<= 1) {
    int v = (t >= d) ? s[t - d] : 0;
    __syncthreads();
    s[t] += v;
    __syncthreads();
  }
  int excl = s[t] - sum;
  for (int i = 0; i < chunk; ++i) {
    int idx = lo + i;
    if (idx < n) { off[idx] = excl; cur[idx] = excl; excl += cnt[idx]; }
  }
  if (t == 1023) off[n] = s[1023];
}

__global__ __launch_bounds__(256) void fill_kernel(const int* __restrict__ r,
                                                   int* __restrict__ cur,
                                                   int* __restrict__ eidx, int ne) {
  int e = blockIdx.x * 256 + threadIdx.x;
  if (e < ne) { int pos = atomicAdd(&cur[r[e]], 1); eidx[pos] = e; }
}

__global__ __launch_bounds__(256) void sort_kernel(const int* __restrict__ off,
                                                   int* __restrict__ eidx, int n) {
  int node = blockIdx.x * 256 + threadIdx.x;
  if (node < n) {
    int b = off[node], e = off[node + 1];
    for (int i = b + 1; i < e; ++i) {
      int v = eidx[i]; int j = i - 1;
      while (j >= b && eidx[j] > v) { eidx[j + 1] = eidx[j]; --j; }
      eidx[j + 1] = v;
    }
  }
}

// ---------------------------------------------------------------------------
// Weight pack: f32 [K][128] row-major -> bf16 fragment-major
//   flat idx = ((cb*KK + kk)*32 + lane)*16 + i  ;  KK = K/32
//   lane: n = lane&15, khalf = lane>>4
//   i<8 : k = kk*32 + khalf*8 + i ; i>=8 : k = kk*32 + 16 + khalf*8 + (i-8)
// ---------------------------------------------------------------------------
__global__ __launch_bounds__(256) void pack_kernel(const float* __restrict__ W,
                                                   __bf16* __restrict__ dst, int K) {
  int step = blockIdx.z;
  const float* w = W + (size_t)step * K * 128;
  __bf16* d = dst + (size_t)step * K * 128;
  int idx = blockIdx.x * 256 + threadIdx.x;
  if (idx >= K * 128) return;
  int i    = idx & 15;
  int lane = (idx >> 4) & 31;
  int rest = idx >> 9;
  int KK = K >> 5;
  int kk = rest % KK;
  int cb = rest / KK;
  int nloc  = lane & 15;
  int khalf = lane >> 4;
  int k = (i < 8) ? (kk * 32 + khalf * 8 + i) : (kk * 32 + 16 + khalf * 8 + (i - 8));
  d[idx] = (__bf16)w[(size_t)k * 128 + cb * 16 + nloc];
}

// ---------------------------------------------------------------------------
// Fused MLP kernel.
//   mode 0: edge   X = [x[s], x[r], io_row]        out: io_row (+residual)
//   mode 1: node   X = [x, mean(ea), mean(eca)]    out: io_row (+residual)
//   mode 2: enc    layer1 = relu(attr@W1+b1) VALU  out: io_row (no residual)
// ---------------------------------------------------------------------------
__global__ __launch_bounds__(WG) void mgn_mlp_kernel(
    int mode, int rows,
    const float* xsrc,              // node states (gather source / self rows)
    float* io,                      // per-row in(old)/out(new) buffer [rows,128]
    const int* __restrict__ idxS, const int* __restrict__ idxR,          // mode 0
    const int* __restrict__ offM, const int* __restrict__ eidxM, const float* ebufM,
    const int* __restrict__ offC, const int* __restrict__ eidxC, const float* ebufC,
    const float* __restrict__ attr, int din, const float* __restrict__ w1raw, // mode 2
    const __bf16* __restrict__ pw1,  // packed layer1 weights (K=384), null in mode 2
    const __bf16* __restrict__ pw2,  // packed layer2 weights (K=128)
    const float* __restrict__ b1, const float* __restrict__ b2,
    const float* __restrict__ gw, const float* __restrict__ bw)
{
  __shared__ __align__(16) __bf16 sX[64 * 392];   // 64x384 bf16 (pad 8) = 49KB
  __shared__ __align__(16) __bf16 sH[64 * 136];   // 64x128 bf16 (pad 8) = 17KB
  __shared__ float sMu[64];
  __shared__ float sRs[64];
  float* sY = (float*)sX;                         // reuse X region: 64x132 f32

  const int tid    = threadIdx.x;
  const int r0     = blockIdx.x * 64;
  const int lane   = tid & 31;
  const int wave   = tid >> 5;
  const int laneLo = lane & 15;
  const int laneHi = lane >> 4;
  const int rb     = wave >> 1;        // row block 0..3
  const int cb0    = (wave & 1) * 4;   // col blocks cb0..cb0+3

  if (mode == 2) {
    // ---- encoder layer 1 (din <= 6) in VALU, straight to sH as bf16 ----
    for (int i = tid; i < 64 * 128; i += WG) {
      int row = i >> 7, c = i & 127;
      int gr = r0 + row; if (gr > rows - 1) gr = rows - 1;
      float h = b1[c];
      for (int k = 0; k < din; ++k) h += attr[(size_t)gr * din + k] * w1raw[k * 128 + c];
      sH[row * 136 + c] = (__bf16)(h > 0.f ? h : 0.f);
    }
  } else {
    // ---- phase 0: build X tile [64 x 384] bf16 in LDS ----
    for (int i = tid; i < 64 * 96; i += WG) {     // 96 float4 per row
      int row = i / 96, c4 = i % 96;
      int gr = r0 + row; if (gr > rows - 1) gr = rows - 1;
      int seg = c4 >> 5, o4 = c4 & 31;
      float4 v;
      if (mode == 0) {
        const float* src;
        if (seg == 0)      src = xsrc + (size_t)idxS[gr] * 128;
        else if (seg == 1) src = xsrc + (size_t)idxR[gr] * 128;
        else               src = io + (size_t)gr * 128;
        v = ((const float4*)src)[o4];
      } else {
        if (seg == 0) {
          v = ((const float4*)(xsrc + (size_t)gr * 128))[o4];
        } else {
          const int*   off  = (seg == 1) ? offM  : offC;
          const int*   eix  = (seg == 1) ? eidxM : eidxC;
          const float* ebuf = (seg == 1) ? ebufM : ebufC;
          int b = off[gr], e = off[gr + 1];
          float ax = 0.f, ay = 0.f, az = 0.f, aw = 0.f;
          for (int j = b; j < e; ++j) {
            float4 t = ((const float4*)(ebuf + (size_t)eix[j] * 128))[o4];
            ax += t.x; ay += t.y; az += t.z; aw += t.w;
          }
          int deg = e - b; if (deg < 1) deg = 1;
          float inv = 1.f / (float)deg;
          v.x = ax * inv; v.y = ay * inv; v.z = az * inv; v.w = aw * inv;
        }
      }
      __bf16* d = sX + row * 392 + c4 * 4;
      d[0] = (__bf16)v.x; d[1] = (__bf16)v.y; d[2] = (__bf16)v.z; d[3] = (__bf16)v.w;
    }
    __syncthreads();
    // ---- layer 1: [64x384] @ [384x128] + b1, ReLU -> sH bf16 ----
    v8f acc[4] = {};
    const __bf16* aBase = sX + (rb * 16 + laneLo) * 392 + laneHi * 8;
    for (int kk = 0; kk < 12; ++kk) {
      v16bf a = ld_frag_a(aBase + kk * 32);
#pragma unroll
      for (int t = 0; t < 4; ++t) {
        v16bf bf = ld_frag_b(pw1 + ((((size_t)(cb0 + t) * 12 + kk) * 32 + lane) << 4));
        acc[t] = __builtin_amdgcn_wmma_f32_16x16x32_bf16(false, a, false, bf,
                                                         (short)0, acc[t], false, false);
      }
    }
#pragma unroll
    for (int t = 0; t < 4; ++t) {
      int n = (cb0 + t) * 16 + laneLo;
      float bv = b1[n];
#pragma unroll
      for (int j = 0; j < 8; ++j) {
        float h = acc[t][j] + bv;
        sH[(rb * 16 + laneHi * 8 + j) * 136 + n] = (__bf16)(h > 0.f ? h : 0.f);
      }
    }
  }
  __syncthreads();

  // ---- layer 2: sH [64x128] @ pw2 [128x128] + b2 -> sY f32 ----
  {
    v8f acc[4] = {};
    const __bf16* aBase = sH + (rb * 16 + laneLo) * 136 + laneHi * 8;
    for (int kk = 0; kk < 4; ++kk) {
      v16bf a = ld_frag_a(aBase + kk * 32);
#pragma unroll
      for (int t = 0; t < 4; ++t) {
        v16bf bf = ld_frag_b(pw2 + ((((size_t)(cb0 + t) * 4 + kk) * 32 + lane) << 4));
        acc[t] = __builtin_amdgcn_wmma_f32_16x16x32_bf16(false, a, false, bf,
                                                         (short)0, acc[t], false, false);
      }
    }
#pragma unroll
    for (int t = 0; t < 4; ++t) {
      int n = (cb0 + t) * 16 + laneLo;
      float bv = b2[n];
#pragma unroll
      for (int j = 0; j < 8; ++j)
        sY[(rb * 16 + laneHi * 8 + j) * 132 + n] = acc[t][j] + bv;
    }
  }
  __syncthreads();

  // ---- LayerNorm stats (population variance, eps=1e-5) ----
  if (tid < 64) {
    const float* yr = sY + tid * 132;
    float s = 0.f, s2 = 0.f;
    for (int c = 0; c < 128; ++c) { float v = yr[c]; s += v; s2 += v * v; }
    float mu  = s * 0.0078125f;
    float var = s2 * 0.0078125f - mu * mu;
    sMu[tid] = mu;
    sRs[tid] = rsqrtf(var + 1e-5f);
  }
  __syncthreads();

  // ---- epilogue: LN * g + be (+ residual), f32 store ----
  for (int i = tid; i < 64 * 32; i += WG) {
    int row = i >> 5, c4 = i & 31;
    int gr = r0 + row;
    if (gr < rows) {
      float mu = sMu[row], rs = sRs[row];
      float4 y  = ((const float4*)(sY + row * 132))[c4];
      float4 gg = ((const float4*)gw)[c4];
      float4 bb = ((const float4*)bw)[c4];
      float4 o;
      o.x = (y.x - mu) * rs * gg.x + bb.x;
      o.y = (y.y - mu) * rs * gg.y + bb.y;
      o.z = (y.z - mu) * rs * gg.z + bb.z;
      o.w = (y.w - mu) * rs * gg.w + bb.w;
      float4* dst = (float4*)(io + (size_t)gr * 128) + c4;
      if (mode != 2) {
        float4 res = *dst;
        o.x += res.x; o.y += res.y; o.z += res.z; o.w += res.w;
      }
      *dst = o;
    }
  }
}

// ---------------------------------------------------------------------------
// Decoder: relu(x@W1+b1) via WMMA, then 128->2 dense in VALU (no LN)
// ---------------------------------------------------------------------------
__global__ __launch_bounds__(WG) void decoder_kernel(
    const float* __restrict__ x,
    const __bf16* __restrict__ pw1,
    const float* __restrict__ b1,
    const float* __restrict__ w2,
    const float* __restrict__ b2,
    float* __restrict__ out, int rows)
{
  __shared__ __align__(16) __bf16 sH[64 * 136];
  __shared__ __align__(16) float  sY[64 * 132];
  const int tid = threadIdx.x, r0 = blockIdx.x * 64;
  const int lane = tid & 31, wave = tid >> 5;
  const int laneLo = lane & 15, laneHi = lane >> 4;
  const int rb = wave >> 1, cb0 = (wave & 1) * 4;

  for (int i = tid; i < 64 * 32; i += WG) {
    int row = i >> 5, c4 = i & 31;
    int gr = r0 + row; if (gr > rows - 1) gr = rows - 1;
    float4 v = ((const float4*)(x + (size_t)gr * 128))[c4];
    __bf16* d = sH + row * 136 + c4 * 4;
    d[0] = (__bf16)v.x; d[1] = (__bf16)v.y; d[2] = (__bf16)v.z; d[3] = (__bf16)v.w;
  }
  __syncthreads();

  v8f acc[4] = {};
  const __bf16* aBase = sH + (rb * 16 + laneLo) * 136 + laneHi * 8;
  for (int kk = 0; kk < 4; ++kk) {
    v16bf a = ld_frag_a(aBase + kk * 32);
#pragma unroll
    for (int t = 0; t < 4; ++t) {
      v16bf bf = ld_frag_b(pw1 + ((((size_t)(cb0 + t) * 4 + kk) * 32 + lane) << 4));
      acc[t] = __builtin_amdgcn_wmma_f32_16x16x32_bf16(false, a, false, bf,
                                                       (short)0, acc[t], false, false);
    }
  }
#pragma unroll
  for (int t = 0; t < 4; ++t) {
    int n = (cb0 + t) * 16 + laneLo;
    float bv = b1[n];
#pragma unroll
    for (int j = 0; j < 8; ++j) {
      float h = acc[t][j] + bv;
      sY[(rb * 16 + laneHi * 8 + j) * 132 + n] = h > 0.f ? h : 0.f;
    }
  }
  __syncthreads();

  if (tid < 128) {
    int row = tid >> 1, o = tid & 1;
    int gr = r0 + row;
    if (gr < rows) {
      const float* h = sY + row * 132;
      float s = b2[o];
      for (int k = 0; k < 128; ++k) s += h[k] * w2[k * 2 + o];
      out[(size_t)gr * 2 + o] = s;
    }
  }
}

// ---------------------------------------------------------------------------
// Host launch
// ---------------------------------------------------------------------------
extern "C" void kernel_launch(void* const* d_in, const int* in_sizes, int n_in,
                              void* d_out, int out_size, void* d_ws, size_t ws_size,
                              hipStream_t stream)
{
  (void)n_in; (void)out_size; (void)ws_size;

  // Inputs in setup_inputs() dict insertion order (leaves: W1,b1,W2,b2,g,be)
  const float* node_attr = (const float*)d_in[0];
  const float* edge_attr = (const float*)d_in[1];
  const float* cont_attr = (const float*)d_in[2];
  const int*   ei        = (const int*)d_in[3];   // [2, EM]
  const int*   eci       = (const int*)d_in[4];   // [2, EC]
  const int N  = in_sizes[0];
  const int EM = in_sizes[1] / 6;
  const int EC = in_sizes[2] / 3;

  const float* enW1n = (const float*)d_in[5];  const float* enB1n = (const float*)d_in[6];
  const float* enW2n = (const float*)d_in[7];  const float* enB2n = (const float*)d_in[8];
  const float* enGn  = (const float*)d_in[9];  const float* enBEn = (const float*)d_in[10];
  const float* enW1m = (const float*)d_in[11]; const float* enB1m = (const float*)d_in[12];
  const float* enW2m = (const float*)d_in[13]; const float* enB2m = (const float*)d_in[14];
  const float* enGm  = (const float*)d_in[15]; const float* enBEm = (const float*)d_in[16];
  const float* enW1c = (const float*)d_in[17]; const float* enB1c = (const float*)d_in[18];
  const float* enW2c = (const float*)d_in[19]; const float* enB2c = (const float*)d_in[20];
  const float* enGc  = (const float*)d_in[21]; const float* enBEc = (const float*)d_in[22];

  const float* pmW1 = (const float*)d_in[23]; const float* pmB1 = (const float*)d_in[24];
  const float* pmW2 = (const float*)d_in[25]; const float* pmB2 = (const float*)d_in[26];
  const float* pmG  = (const float*)d_in[27]; const float* pmBE = (const float*)d_in[28];
  const float* pcW1 = (const float*)d_in[29]; const float* pcB1 = (const float*)d_in[30];
  const float* pcW2 = (const float*)d_in[31]; const float* pcB2 = (const float*)d_in[32];
  const float* pcG  = (const float*)d_in[33]; const float* pcBE = (const float*)d_in[34];
  const float* pnW1 = (const float*)d_in[35]; const float* pnB1 = (const float*)d_in[36];
  const float* pnW2 = (const float*)d_in[37]; const float* pnB2 = (const float*)d_in[38];
  const float* pnG  = (const float*)d_in[39]; const float* pnBE = (const float*)d_in[40];

  const float* dW1 = (const float*)d_in[41]; const float* dB1 = (const float*)d_in[42];
  const float* dW2 = (const float*)d_in[43]; const float* dB2 = (const float*)d_in[44];

  const int S = in_sizes[24] / 128;   // proc mesh b1 stacked [S,128]

  // ---- workspace carve ----
  char* p = (char*)d_ws;
  auto carve = [&](size_t bytes) -> char* {
    char* r = p; p += (bytes + 255) & ~(size_t)255; return r;
  };
  float* x    = (float*)carve((size_t)N  * 128 * 4);
  float* ea   = (float*)carve((size_t)EM * 128 * 4);
  float* eca  = (float*)carve((size_t)EC * 128 * 4);
  int* cntM = (int*)carve((size_t)N * 4);
  int* offM = (int*)carve((size_t)(N + 1) * 4);
  int* curM = (int*)carve((size_t)(N + 1) * 4);
  int* eixM = (int*)carve((size_t)EM * 4);
  int* cntC = (int*)carve((size_t)N * 4);
  int* offC = (int*)carve((size_t)(N + 1) * 4);
  int* curC = (int*)carve((size_t)(N + 1) * 4);
  int* eixC = (int*)carve((size_t)EC * 4);
  __bf16* w1m  = (__bf16*)carve((size_t)S * 384 * 128 * 2);
  __bf16* w2m  = (__bf16*)carve((size_t)S * 128 * 128 * 2);
  __bf16* w1c  = (__bf16*)carve((size_t)S * 384 * 128 * 2);
  __bf16* w2c  = (__bf16*)carve((size_t)S * 128 * 128 * 2);
  __bf16* w1n  = (__bf16*)carve((size_t)S * 384 * 128 * 2);
  __bf16* w2n  = (__bf16*)carve((size_t)S * 128 * 128 * 2);
  __bf16* w2en = (__bf16*)carve((size_t)128 * 128 * 2);
  __bf16* w2em = (__bf16*)carve((size_t)128 * 128 * 2);
  __bf16* w2ec = (__bf16*)carve((size_t)128 * 128 * 2);
  __bf16* w1d  = (__bf16*)carve((size_t)128 * 128 * 2);

  // ---- CSR build (deterministic) ----
  hipMemsetAsync(cntM, 0, (size_t)N * 4, stream);
  hipMemsetAsync(cntC, 0, (size_t)N * 4, stream);
  count_kernel<<<(EM + 255) / 256, 256, 0, stream>>>(ei + EM, cntM, EM);
  count_kernel<<<(EC + 255) / 256, 256, 0, stream>>>(eci + EC, cntC, EC);
  int chunk = (N + 1023) / 1024;
  scan_kernel<<<1, 1024, 0, stream>>>(cntM, offM, curM, N, chunk);
  scan_kernel<<<1, 1024, 0, stream>>>(cntC, offC, curC, N, chunk);
  fill_kernel<<<(EM + 255) / 256, 256, 0, stream>>>(ei + EM, curM, eixM, EM);
  fill_kernel<<<(EC + 255) / 256, 256, 0, stream>>>(eci + EC, curC, eixC, EC);
  sort_kernel<<<(N + 255) / 256, 256, 0, stream>>>(offM, eixM, N);
  sort_kernel<<<(N + 255) / 256, 256, 0, stream>>>(offC, eixC, N);

  // ---- weight packing (f32 -> fragment-major bf16) ----
  dim3 g384((384 * 128 + 255) / 256, 1, S);
  dim3 g128((128 * 128 + 255) / 256, 1, S);
  dim3 g1(64, 1, 1);
  pack_kernel<<<g384, 256, 0, stream>>>(pmW1, w1m, 384);
  pack_kernel<<<g128, 256, 0, stream>>>(pmW2, w2m, 128);
  pack_kernel<<<g384, 256, 0, stream>>>(pcW1, w1c, 384);
  pack_kernel<<<g128, 256, 0, stream>>>(pcW2, w2c, 128);
  pack_kernel<<<g384, 256, 0, stream>>>(pnW1, w1n, 384);
  pack_kernel<<<g128, 256, 0, stream>>>(pnW2, w2n, 128);
  pack_kernel<<<g1, 256, 0, stream>>>(enW2n, w2en, 128);
  pack_kernel<<<g1, 256, 0, stream>>>(enW2m, w2em, 128);
  pack_kernel<<<g1, 256, 0, stream>>>(enW2c, w2ec, 128);
  pack_kernel<<<g1, 256, 0, stream>>>(dW1, w1d, 128);

  const int gN = (N + 63) / 64, gEM = (EM + 63) / 64, gEC = (EC + 63) / 64;

  // ---- encoders ----
  mgn_mlp_kernel<<<gN, WG, 0, stream>>>(2, N, nullptr, x,
      nullptr, nullptr, nullptr, nullptr, nullptr, nullptr, nullptr, nullptr,
      node_attr, 1, enW1n, nullptr, w2en, enB1n, enB2n, enGn, enBEn);
  mgn_mlp_kernel<<<gEM, WG, 0, stream>>>(2, EM, nullptr, ea,
      nullptr, nullptr, nullptr, nullptr, nullptr, nullptr, nullptr, nullptr,
      edge_attr, 6, enW1m, nullptr, w2em, enB1m, enB2m, enGm, enBEm);
  mgn_mlp_kernel<<<gEC, WG, 0, stream>>>(2, EC, nullptr, eca,
      nullptr, nullptr, nullptr, nullptr, nullptr, nullptr, nullptr, nullptr,
      cont_attr, 3, enW1c, nullptr, w2ec, enB1c, enB2c, enGc, enBEc);

  // ---- processor steps ----
  for (int i = 0; i < S; ++i) {
    const size_t o1 = (size_t)i * 384 * 128, o2 = (size_t)i * 128 * 128;
    mgn_mlp_kernel<<<gEM, WG, 0, stream>>>(0, EM, x, ea,
        ei, ei + EM, nullptr, nullptr, nullptr, nullptr, nullptr, nullptr,
        nullptr, 0, nullptr, w1m + o1, w2m + o2,
        pmB1 + i * 128, pmB2 + i * 128, pmG + i * 128, pmBE + i * 128);
    mgn_mlp_kernel<<<gEC, WG, 0, stream>>>(0, EC, x, eca,
        eci, eci + EC, nullptr, nullptr, nullptr, nullptr, nullptr, nullptr,
        nullptr, 0, nullptr, w1c + o1, w2c + o2,
        pcB1 + i * 128, pcB2 + i * 128, pcG + i * 128, pcBE + i * 128);
    mgn_mlp_kernel<<<gN, WG, 0, stream>>>(1, N, x, x,
        nullptr, nullptr, offM, eixM, ea, offC, eixC, eca,
        nullptr, 0, nullptr, w1n + o1, w2n + o2,
        pnB1 + i * 128, pnB2 + i * 128, pnG + i * 128, pnBE + i * 128);
  }

  // ---- decoder ----
  decoder_kernel<<<gN, WG, 0, stream>>>(x, w1d, dB1, dW2, dB2, (float*)d_out, N);
}
